// LIIF_78202764525587
// MI455X (gfx1250) — compile-verified, matrix operand order checked by reading, not run on hardware
//
#include <hip/hip_runtime.h>

typedef __attribute__((ext_vector_type(16))) _Float16 v16h;
typedef __attribute__((ext_vector_type(8)))  float    v8f;

#define H_LR 128
#define W_LR 128
#define NQ   (512*512)
#define CF   64
#define K0P  608          // 580 padded to multiple of 32
#define NCH0 19           // K0P / 32 chunks
#define NH   256
#define SPB  64           // query samples per block
#define ROWS 256          // SPB * 4 ensemble branches = GEMM rows
#define MT   16           // m-tiles (ROWS/16)
#define THREADS 512       // 16 waves
#define PADW 130          // zero-bordered feature map width

// ---------------- dynamic LDS layout (bytes, 16B aligned) ----------------
#define LDS_SH    0u
#define LDS_SAC   (LDS_SH + ROWS*NH*2u)          // 131072
#define LDS_AREA  (LDS_SAC + 2u*ROWS*32u*2u)     // +32768 = 163840
#define LDS_PRED  (LDS_AREA + ROWS*4u)           // +1024  = 164864
#define LDS_C0    (LDS_PRED + ROWS*4u*4u)        // +4096  = 168960
#define LDS_C1    (LDS_C0 + SPB*4u)              // +256   = 169216
#define LDS_TOTAL (LDS_C1 + SPB*4u)              // +256   = 169472

// ---------------- workspace layout (bytes, 16B aligned) ----------------
#define OFF_FEAT 0u
#define SZ_FEAT  (PADW*PADW*CF*2u)        // 2,163,200  layout [y][x][c] f16
#define OFF_W0T  (OFF_FEAT + SZ_FEAT)
#define SZ_W0T   (NH*K0P*2u)              //   311,296  [n][k'] K-permuted
#define OFF_W1T  (OFF_W0T + SZ_W0T)
#define SZ_WHT   (NH*NH*2u)               //   131,072
#define OFF_W2T  (OFF_W1T + SZ_WHT)
#define OFF_W3T  (OFF_W2T + SZ_WHT)
#define OFF_W4T  (OFF_W3T + SZ_WHT)       // + 8,192

union FragU    { v16h h; uint4 u[2]; };
union HalfPack { _Float16 h[8]; uint4 u; };

// Fragment load for V_WMMA_F32_16X16X32_F16 (ISA 7.12.2): lane holds row l&15;
// halves 0..7 = K[k0 + 8*(l>=16)..], halves 8..15 = K[k0+16 + 8*(l>=16)..].
__device__ __forceinline__ v16h load_frag(const _Float16* base, int stride, int k0, int lane) {
  const _Float16* p = base + (lane & 15) * stride + k0 + ((lane >> 4) & 1) * 8;
  FragU f;
  f.u[0] = *(const uint4*)(p);
  f.u[1] = *(const uint4*)(p + 16);
  return f.h;
}

__device__ __forceinline__ v8f wmma16(v16h a, v16h b, v8f c) {
  return __builtin_amdgcn_wmma_f32_16x16x32_f16(false, a, false, b, (short)0, c, false, false);
}

// CDNA5 async global->LDS copy (ISA 10.4 / 15.18.3), tracked by ASYNCcnt.
// Flat/generic LDS pointers carry the wave-relative LDS byte address in the
// low 32 bits (ISA 10.2 aperture mapping), which is what VDST expects.
__device__ __forceinline__ void async_ld_b128(void* ldst, const void* gsrc) {
  unsigned l = (unsigned)(uintptr_t)ldst;
  unsigned long long g = (unsigned long long)(uintptr_t)gsrc;
  asm volatile("global_load_async_to_lds_b128 %0, %1, off"
               :: "v"(l), "v"(g) : "memory");
}
__device__ __forceinline__ void wait_async0() {
#if __has_builtin(__builtin_amdgcn_s_wait_asynccnt)
  __builtin_amdgcn_s_wait_asynccnt(0);
#else
  asm volatile("s_wait_asynccnt 0" ::: "memory");
#endif
}

// ---------------- bicubic (align_corners=False, border, a=-0.75) ----------------
__device__ __forceinline__ float cub1(float x) {
  const float A = -0.75f;
  return ((A + 2.f) * x - (A + 3.f)) * x * x + 1.f;
}
__device__ __forceinline__ float cub2(float x) {
  const float A = -0.75f;
  return ((A * x - 5.f * A) * x + 8.f * A) * x - 4.f * A;
}
__device__ float bicubic_border(const float* __restrict__ img, int ch, float gy, float gx) {
  float fy = ((gy + 1.f) * (float)H_LR - 1.f) * 0.5f;
  float fx = ((gx + 1.f) * (float)W_LR - 1.f) * 0.5f;
  float y0 = floorf(fy), x0 = floorf(fx);
  float ty = fy - y0, tx = fx - x0;
  float wy[4] = { cub2(ty + 1.f), cub1(ty), cub1(1.f - ty), cub2(2.f - ty) };
  float wx[4] = { cub2(tx + 1.f), cub1(tx), cub1(1.f - tx), cub2(2.f - tx) };
  int yi = (int)y0, xi = (int)x0;
  const float* base = img + ch * H_LR * W_LR;
  float acc = 0.f;
#pragma unroll
  for (int i = 0; i < 4; ++i) {
    int yy = yi + i - 1; yy = yy < 0 ? 0 : (yy > H_LR - 1 ? H_LR - 1 : yy);
#pragma unroll
    for (int j = 0; j < 4; ++j) {
      int xx = xi + j - 1; xx = xx < 0 ? 0 : (xx > W_LR - 1 ? W_LR - 1 : xx);
      acc += wy[i] * wx[j] * base[yy * W_LR + xx];
    }
  }
  return acc;
}

// ---------------- encoder conv: 3->64, SAME; output [y][x][c] f16, zero border ----------------
__global__ __launch_bounds__(128) void conv_feat(const float* __restrict__ inp,
                                                 const float* __restrict__ cw,
                                                 const float* __restrict__ cb,
                                                 _Float16* __restrict__ featp) {
  __shared__ float sw[CF * 27];
  __shared__ float sb[CF];
  for (int i = threadIdx.x; i < CF * 27; i += 128) sw[i] = cw[i];
  for (int i = threadIdx.x; i < CF; i += 128) sb[i] = cb[i];
  __syncthreads();
  int idx = blockIdx.x * 128 + threadIdx.x;
  if (idx >= PADW * PADW) return;
  int yp = idx / PADW, xp = idx - yp * PADW;
  _Float16* dst = featp + (size_t)idx * CF;
  if (yp == 0 || yp == PADW - 1 || xp == 0 || xp == PADW - 1) {
    for (int o = 0; o < CF; ++o) dst[o] = (_Float16)0.f;
    return;
  }
  int y = yp - 1, x = xp - 1;
  float patch[27];
#pragma unroll
  for (int c = 0; c < 3; ++c)
#pragma unroll
    for (int ky = 0; ky < 3; ++ky)
#pragma unroll
      for (int kx = 0; kx < 3; ++kx) {
        int yy = y + ky - 1, xx = x + kx - 1;
        float v = 0.f;
        if (yy >= 0 && yy < H_LR && xx >= 0 && xx < W_LR)
          v = inp[(c * H_LR + yy) * W_LR + xx];
        patch[c * 9 + ky * 3 + kx] = v;
      }
  for (int o = 0; o < CF; ++o) {
    float acc = sb[o];
#pragma unroll
    for (int t = 0; t < 27; ++t) acc += sw[o * 27 + t] * patch[t];
    dst[o] = (_Float16)acc;
  }
}

// ---------------- layer-0 weights: f32 [580][256] -> f16 [256][608], K permuted ----------------
// k' = rr*64 + c  maps original k = c*9 + rr  (rr = ky*3+kx); k' 576..579 pass-through.
__global__ void transpose_pad_l0(const float* __restrict__ src, _Float16* __restrict__ dst) {
  int idx = blockIdx.x * blockDim.x + threadIdx.x;
  if (idx >= NH * K0P) return;
  int n = idx / K0P, kp = idx - n * K0P;
  float v = 0.f;
  if (kp < 576) {
    int c = kp & 63, rr = kp >> 6;
    v = src[(c * 9 + rr) * NH + n];
  } else if (kp < 580) {
    v = src[kp * NH + n];
  }
  dst[n * K0P + kp] = (_Float16)v;
}

// ---------------- generic weight prep: f32 [K][N] -> f16 transposed/padded [NP][KP] ----------------
__global__ void transpose_pad(const float* __restrict__ src, _Float16* __restrict__ dst,
                              int K, int N, int KP, int NP) {
  int idx = blockIdx.x * blockDim.x + threadIdx.x;
  if (idx >= NP * KP) return;
  int n = idx / KP, k = idx - n * KP;
  float v = (k < K && n < N) ? src[k * N + n] : 0.f;
  dst[n * KP + k] = (_Float16)v;
}

// ---------------- in-place hidden layer: [256 x 256] @ [256 x 256], ReLU ----------------
// Wave owns 1 n-tile x all 16 m-tiles: every B fragment loaded exactly once per block.
__device__ __forceinline__ void hidden_layer(_Float16* sH, const _Float16* __restrict__ wt,
                                             const float* __restrict__ bias,
                                             int wave, int lane) {
  v8f acc[MT] = {};
  const _Float16* wbase = wt + wave * 16 * NH;
  for (int kk = 0; kk < NH; kk += 32) {
    v16h b = load_frag(wbase, NH, kk, lane);
#pragma unroll
    for (int m = 0; m < MT; ++m) {
      v16h a = load_frag(sH + m * 16 * NH, NH, kk, lane);
      acc[m] = wmma16(a, b, acc[m]);
    }
  }
  __syncthreads();          // all reads of sH complete -> safe to overwrite in place
  int n = wave * 16 + (lane & 15);
  float bv = bias[n];
#pragma unroll
  for (int m = 0; m < MT; ++m) {
    int mrow0 = m * 16 + ((lane >> 4) << 3);
#pragma unroll
    for (int r = 0; r < 8; ++r) {
      float v = acc[m][r] + bv;
      v = v > 0.f ? v : 0.f;
      sH[(mrow0 + r) * NH + n] = (_Float16)v;
    }
  }
  __syncthreads();
}

// ---------------- main: async chunked gather + 5-layer WMMA MLP + ensemble + bicubic ----------------
__global__ __launch_bounds__(THREADS) void liif_main(
    const float* __restrict__ inp, const float* __restrict__ coord,
    const float* __restrict__ cell,
    const float* __restrict__ b0, const float* __restrict__ b1,
    const float* __restrict__ b2, const float* __restrict__ b3,
    const float* __restrict__ b4,
    const _Float16* __restrict__ featp,
    const _Float16* __restrict__ w0t, const _Float16* __restrict__ w1t,
    const _Float16* __restrict__ w2t, const _Float16* __restrict__ w3t,
    const _Float16* __restrict__ w4t,
    float* __restrict__ out) {
  extern __shared__ __align__(16) char smem[];
  _Float16* sH    = (_Float16*)(smem + LDS_SH);     // 128 KB hidden buffer (in-place)
  _Float16* sAc   = (_Float16*)(smem + LDS_SAC);    // 32 KB layer-0 K-chunk ping-pong
  float*    sArea = (float*)(smem + LDS_AREA);
  float*    sPred = (float*)(smem + LDS_PRED);      // [row][4]
  float*    sC0   = (float*)(smem + LDS_C0);
  float*    sC1   = (float*)(smem + LDS_C1);

  const int tid  = threadIdx.x;
  const int lane = tid & 31;
  const int wave = tid >> 5;

  // ---- Phase 0: per-row metadata, kept in registers (2 threads per row) ----
  const int row = tid >> 1;          // 0..255 : row = sample*4 + branch
  const int sub = tid & 1;
  int s = row >> 2;
  int j = row & 3;                   // (vx,vy) = ((-1,-1),(-1,1),(1,-1),(1,1))
  int p = blockIdx.x * SPB + s;
  float c0v = coord[2 * p + 0];
  float c1v = coord[2 * p + 1];
  float ce0 = cell[2 * p + 0] * (float)H_LR;
  float ce1 = cell[2 * p + 1] * (float)W_LR;
  float vx = (j & 2) ? 1.f : -1.f;
  float vy = (j & 1) ? 1.f : -1.f;
  const float rx = 1.f / (float)H_LR, ry = 1.f / (float)W_LR;
  float cc0 = fminf(fmaxf(c0v + vx * rx + 1e-6f, -1.f + 1e-6f), 1.f - 1e-6f);
  float cc1 = fminf(fmaxf(c1v + vy * ry + 1e-6f, -1.f + 1e-6f), 1.f - 1e-6f);
  int iy = (int)rintf(((cc0 + 1.f) * (float)H_LR - 1.f) * 0.5f);   // round half-even
  int ix = (int)rintf(((cc1 + 1.f) * (float)W_LR - 1.f) * 0.5f);
  iy = iy < 0 ? 0 : (iy > H_LR - 1 ? H_LR - 1 : iy);
  ix = ix < 0 ? 0 : (ix > W_LR - 1 ? W_LR - 1 : ix);
  float qc0 = -1.f + (2.f * iy + 1.f) * rx;
  float qc1 = -1.f + (2.f * ix + 1.f) * ry;
  float rel0 = (c0v - qc0) * (float)H_LR;
  float rel1 = (c1v - qc1) * (float)W_LR;
  const int pos = iy * PADW + ix;
  if (sub == 0) {
    sArea[row] = fabsf(rel0 * rel1) + 1e-9f;
    if (j == 0) { sC0[s] = c0v; sC1[s] = c1v; }
  }
  HalfPack relPack;
#pragma unroll
  for (int q = 0; q < 8; ++q) relPack.h[q] = (_Float16)0.f;
  relPack.h[0] = (_Float16)rel0;
  relPack.h[1] = (_Float16)rel1;
  relPack.h[2] = (_Float16)ce0;
  relPack.h[3] = (_Float16)ce1;
  // warm L2 with this wave's layer-0 weight tile
  __builtin_prefetch((const char*)(w0t + wave * 16 * K0P) + (lane << 7), 0, 1);

  // chunk i covers k' = 32i..32i+31; for k'<576: rr = i>>1 (ky=rr/3, kx=rr%3),
  // channels c = (i&1)*32 + sub*16 .. +15, contiguous in featp[y][x][c].
  // Feature chunks go straight global->LDS via async DMA (ASYNCcnt).
  auto issue_chunk = [&](int i, int buf) {
    _Float16* d = sAc + buf * (ROWS * 32) + row * 32 + (sub << 4);
    if (i == NCH0 - 1) {             // rel/cell + zero padding chunk (register data)
      uint4 z; z.x = 0; z.y = 0; z.z = 0; z.w = 0;
      *(uint4*)d = (sub == 0) ? relPack.u : z;
      *(uint4*)(d + 8) = z;
    } else {
      int rr = i >> 1;
      int dy = rr / 3, dx = rr - dy * 3;
      int c0i = ((i & 1) << 5) + (sub << 4);
      const _Float16* gp = featp + (size_t)(pos + dy * PADW + dx) * CF + c0i;
      async_ld_b128(d, gp);
      async_ld_b128(d + 8, gp + 8);
    }
  };

  // ---- Layer 0: [256 x 608] @ [608 x 256] with async chunk-pipelined A ----
  issue_chunk(0, 0);
  wait_async0();
  __syncthreads();

  {
    v8f acc[MT] = {};
    const _Float16* wbase = w0t + wave * 16 * K0P;
    for (int i = 0; i < NCH0; ++i) {
      int cur = i & 1;
      bool have = (i + 1) < NCH0;
      if (have) issue_chunk(i + 1, 1 - cur);   // async DMA overlaps the WMMA burst
      v16h b = load_frag(wbase, K0P, i * 32, lane);
      const _Float16* abase = sAc + cur * (ROWS * 32);
#pragma unroll
      for (int m = 0; m < MT; ++m) {
        v16h a = load_frag(abase + m * 16 * 32, 32, 0, lane);
        acc[m] = wmma16(a, b, acc[m]);
      }
      if (have) wait_async0();                 // fence own async writes before barrier
      __syncthreads();
    }
    // epilogue -> sH (ReLU)
    int n = wave * 16 + (lane & 15);
    float bv = b0[n];
#pragma unroll
    for (int m = 0; m < MT; ++m) {
      int mrow0 = m * 16 + ((lane >> 4) << 3);
#pragma unroll
      for (int r = 0; r < 8; ++r) {
        float v = acc[m][r] + bv;
        v = v > 0.f ? v : 0.f;
        sH[(mrow0 + r) * NH + n] = (_Float16)v;
      }
    }
  }
  __syncthreads();

  hidden_layer(sH, w1t, b1, wave, lane);
  hidden_layer(sH, w2t, b2, wave, lane);
  hidden_layer(sH, w3t, b3, wave, lane);

  // ---- Layer 4: [256 x 256] @ [256 x 16]; wave w handles m-tile w ----
  {
    v8f acc4 = {};
    const _Float16* A = sH + wave * 16 * NH;
    for (int kk = 0; kk < NH; kk += 32) {
      v16h a = load_frag(A, NH, kk, lane);
      v16h b = load_frag(w4t, NH, kk, lane);
      acc4 = wmma16(a, b, acc4);
    }
    int n = lane & 15;
    if (n < 3) {
      float bv = b4[n];
      int mrow0 = wave * 16 + ((lane >> 4) << 3);
#pragma unroll
      for (int r = 0; r < 8; ++r) sPred[(mrow0 + r) * 4 + n] = acc4[r] + bv;
    }
  }
  __syncthreads();

  // ---- Local-ensemble combine (diagonal area swap) + bicubic residual ----
  if (tid < SPB * 3) {
    int ss = tid / 3, ch = tid - ss * 3;
    float a0 = sArea[ss * 4 + 0], a1 = sArea[ss * 4 + 1];
    float a2 = sArea[ss * 4 + 2], a3 = sArea[ss * 4 + 3];
    float tot = a0 + a1 + a2 + a3;
    float v = sPred[(ss * 4 + 0) * 4 + ch] * a3 + sPred[(ss * 4 + 1) * 4 + ch] * a2 +
              sPred[(ss * 4 + 2) * 4 + ch] * a1 + sPred[(ss * 4 + 3) * 4 + ch] * a0;
    v /= tot;
    v += bicubic_border(inp, ch, sC0[ss], sC1[ss]);
    int pp = blockIdx.x * SPB + ss;
    out[ch * NQ + pp] = v;
  }
}

extern "C" void kernel_launch(void* const* d_in, const int* in_sizes, int n_in,
                              void* d_out, int out_size, void* d_ws, size_t ws_size,
                              hipStream_t stream) {
  const float* inp    = (const float*)d_in[0];
  const float* coord  = (const float*)d_in[1];
  const float* cell   = (const float*)d_in[2];
  const float* conv_w = (const float*)d_in[3];
  const float* conv_b = (const float*)d_in[4];
  const float* w0 = (const float*)d_in[5];
  const float* b0 = (const float*)d_in[6];
  const float* w1 = (const float*)d_in[7];
  const float* b1 = (const float*)d_in[8];
  const float* w2 = (const float*)d_in[9];
  const float* b2 = (const float*)d_in[10];
  const float* w3 = (const float*)d_in[11];
  const float* b3 = (const float*)d_in[12];
  const float* w4 = (const float*)d_in[13];
  const float* b4 = (const float*)d_in[14];
  char* ws = (char*)d_ws;
  _Float16* featp = (_Float16*)(ws + OFF_FEAT);
  _Float16* w0t   = (_Float16*)(ws + OFF_W0T);
  _Float16* w1t   = (_Float16*)(ws + OFF_W1T);
  _Float16* w2t   = (_Float16*)(ws + OFF_W2T);
  _Float16* w3t   = (_Float16*)(ws + OFF_W3T);
  _Float16* w4t   = (_Float16*)(ws + OFF_W4T);
  float* out = (float*)d_out;

  (void)hipFuncSetAttribute((const void*)liif_main,
                            hipFuncAttributeMaxDynamicSharedMemorySize, LDS_TOTAL);

  conv_feat<<<(PADW * PADW + 127) / 128, 128, 0, stream>>>(inp, conv_w, conv_b, featp);
  transpose_pad_l0<<<(NH * K0P + 255) / 256, 256, 0, stream>>>(w0, w0t);
  transpose_pad<<<(NH * NH + 255) / 256, 256, 0, stream>>>(w1, w1t, NH, NH, NH, NH);
  transpose_pad<<<(NH * NH + 255) / 256, 256, 0, stream>>>(w2, w2t, NH, NH, NH, NH);
  transpose_pad<<<(NH * NH + 255) / 256, 256, 0, stream>>>(w3, w3t, NH, NH, NH, NH);
  transpose_pad<<<(16 * NH + 255) / 256, 256, 0, stream>>>(w4, w4t, NH, 3, NH, 16);
  liif_main<<<NQ / SPB, THREADS, LDS_TOTAL, stream>>>(inp, coord, cell, b0, b1, b2, b3, b4,
                                                      featp, w0t, w1t, w2t, w3t, w4t, out);
}
